// MultiHeadAttention_60739427500577
// MI455X (gfx1250) — matrix operand order repaired and checked by reading, not compiled, not run on hardware
//
#include <hip/hip_runtime.h>
#include <hip/hip_bf16.h>
#include <stdint.h>

// ---------------------------------------------------------------------------
// Problem constants (B=4, T=2048, D=1024, H=16, DH=64)
// ---------------------------------------------------------------------------
#define BB   4
#define TT   2048
#define DD   1024
#define HH   16
#define DH   64
#define MM   (BB * TT)

typedef __attribute__((ext_vector_type(16))) __bf16 v16bf;
typedef __attribute__((ext_vector_type(8)))  float  v8f;
typedef __attribute__((ext_vector_type(4)))  unsigned int u32x4;
typedef __attribute__((ext_vector_type(8)))  unsigned int u32x8;

union FragBF {
    v16bf v;
    uint4 u[2];
};

#define WMMA_BF16(A, Bf, C) \
    __builtin_amdgcn_wmma_f32_16x16x32_bf16(false, (A), false, (Bf), (short)0, (C), false, false)

// LDS row stride (elements): multiple of 8 -> every 8-elem chunk is 16B aligned
#define LDS_K 72

// ---------------------------------------------------------------------------
// CDNA5 async staging: global -> LDS without a VGPR round trip (ASYNCcnt)
// ---------------------------------------------------------------------------
__device__ __forceinline__ void async_b128(const __bf16* g, __bf16* l) {
    unsigned lds_off = (unsigned)(size_t)(void*)l;   // LDS aperture: addr[31:0]
    asm volatile("global_load_async_to_lds_b128 %0, %1, off"
                 :: "v"(lds_off), "v"(g) : "memory");
}
__device__ __forceinline__ void wait_async0() {
    asm volatile("s_wait_asynccnt 0x0" ::: "memory");
}

// ---------------------------------------------------------------------------
// CDNA5 Tensor Data Mover: 2D 64x64 tile load, rows padded to LDS stride 72
// (pad_interval=32 dwords = one 64-elem bf16 row, pad_amount=4 dwords = 8 elems)
// ---------------------------------------------------------------------------
__device__ __forceinline__ void tdm_load_2d_64x64(unsigned lds_addr,
                                                  unsigned long long gaddr,
                                                  unsigned dim0, unsigned dim1,
                                                  unsigned stride0) {
    u32x4 g0;
    g0[0] = 1u;                                   // count = 1 valid descriptor
    g0[1] = lds_addr;                             // lds_addr (bytes)
    g0[2] = (unsigned)gaddr;                      // global_addr[31:0]
    g0[3] = (unsigned)(gaddr >> 32) | (2u << 30); // global_addr[56:32] | type=2
    u32x8 g1;
    g1[0] = (1u << 16)        // data_size = 2 bytes
          | (1u << 20)        // pad_enable
          | (4u << 22)        // pad_interval: 32 dwords
          | (3u << 25);       // pad_amount: 4 dwords  -> LDS row stride 72 elems
    g1[1] = (dim0 & 0xFFFFu) << 16;                       // tensor_dim0[15:0]
    g1[2] = (dim0 >> 16) | ((dim1 & 0xFFFFu) << 16);      // dim0[31:16]|dim1[15:0]
    g1[3] = (dim1 >> 16) | (64u << 16);                   // dim1[31:16]|tile_dim0=64
    g1[4] = 64u;                                          // tile_dim1=64, tile_dim2=0
    g1[5] = stride0;                                      // tensor_dim0_stride lo32
    g1[6] = 0u;
    g1[7] = 0u;
    asm volatile("tensor_load_to_lds %0, %1" :: "s"(g0), "s"(g1) : "memory");
}

__device__ __forceinline__ float rowmax16(float v) {
    v = fmaxf(v, __shfl_xor(v, 1, 32));
    v = fmaxf(v, __shfl_xor(v, 2, 32));
    v = fmaxf(v, __shfl_xor(v, 4, 32));
    v = fmaxf(v, __shfl_xor(v, 8, 32));
    return v;
}
__device__ __forceinline__ float rowsum16(float v) {
    v += __shfl_xor(v, 1, 32);
    v += __shfl_xor(v, 2, 32);
    v += __shfl_xor(v, 4, 32);
    v += __shfl_xor(v, 8, 32);
    return v;
}

// ---------------------------------------------------------------------------
// Kernel 0a: fp32 -> bf16 flat conversion (x)
// ---------------------------------------------------------------------------
__global__ __launch_bounds__(256) void cvt_f32_bf16(const float* __restrict__ in,
                                                    __bf16* __restrict__ out, int n) {
    int idx = (blockIdx.x * 256 + threadIdx.x) * 4;
    if (idx < n) {
        float4 f = *(const float4*)(in + idx);
        out[idx + 0] = (__bf16)f.x;
        out[idx + 1] = (__bf16)f.y;
        out[idx + 2] = (__bf16)f.z;
        out[idx + 3] = (__bf16)f.w;
    }
}

// Kernel 0b: fp32 [K][N] -> bf16 transposed [N][K] (weights), 16x16 tiles
__global__ __launch_bounds__(256) void cvt_transpose(const float* __restrict__ w,
                                                     __bf16* __restrict__ wt) {
    const int n0 = blockIdx.x * 16, k0 = blockIdx.y * 16;
    const int tx = threadIdx.x & 15, ty = threadIdx.x >> 4;
    wt[(size_t)(n0 + tx) * DD + k0 + ty] = (__bf16)w[(size_t)(k0 + ty) * DD + n0 + tx];
}

// ---------------------------------------------------------------------------
// GEMM core: C[128x64] tile, BK=64, 256 threads = 8 waves (4m x 2n), each wave
// a 32x32 register tile. Double-buffered async global->LDS staging.
// A: row-major [M][1024] bf16.  Wt: pre-transposed [N][1024] bf16.
// ---------------------------------------------------------------------------
#define AS_BUF (128 * LDS_K)
#define BS_BUF (64 * LDS_K)

__device__ __forceinline__ void gemm_stage(const __bf16* __restrict__ Ag,
                                           const __bf16* __restrict__ Wt,
                                           __bf16* As, __bf16* Bs,
                                           int m0, int n0, int k0, int p) {
    const int tid = threadIdx.x;
    const int a_row = tid >> 1, a_kg = (tid & 1) * 32;   // 128 rows x 64 k
    const int b_row = tid >> 2, b_kg = (tid & 3) * 16;   // 64 rows  x 64 k
    const __bf16* ga = Ag + (size_t)(m0 + a_row) * DD + k0 + a_kg;
    __bf16* la = As + p * AS_BUF + a_row * LDS_K + a_kg;
    async_b128(ga,      la);
    async_b128(ga + 8,  la + 8);
    async_b128(ga + 16, la + 16);
    async_b128(ga + 24, la + 24);
    const __bf16* gb = Wt + (size_t)(n0 + b_row) * DD + k0 + b_kg;
    __bf16* lb = Bs + p * BS_BUF + b_row * LDS_K + b_kg;
    async_b128(gb,     lb);
    async_b128(gb + 8, lb + 8);
}

__device__ __forceinline__ void gemm_core(const __bf16* __restrict__ Ag,
                                          const __bf16* __restrict__ Wt,
                                          __bf16* As, __bf16* Bs,
                                          int m0, int n0, v8f acc[2][2]) {
    const int lane  = threadIdx.x & 31;
    const int wid   = threadIdx.x >> 5;
    const int wm    = wid >> 1, wn = wid & 1;
    const int l15   = lane & 15, khalf = lane >> 4;

    gemm_stage(Ag, Wt, As, Bs, m0, n0, 0, 0);
    wait_async0();
    __syncthreads();

    for (int k0 = 0; k0 < DD; k0 += 64) {
        const int p = (k0 >> 6) & 1;
        if (k0 + 64 < DD)                       // prefetch next K-step (overlaps WMMA)
            gemm_stage(Ag, Wt, As, Bs, m0, n0, k0 + 64, p ^ 1);

        const __bf16* Ab = As + p * AS_BUF;
        const __bf16* Bb = Bs + p * BS_BUF;
        #pragma unroll
        for (int kc = 0; kc < 2; ++kc) {
            FragBF a0, a1, b0, b1;
            const __bf16* ar0 = Ab + (wm * 32 + l15)      * LDS_K + kc * 32;
            const __bf16* ar1 = Ab + (wm * 32 + 16 + l15) * LDS_K + kc * 32;
            a0.u[0] = *(const uint4*)(ar0 + khalf * 8);
            a0.u[1] = *(const uint4*)(ar0 + 16 + khalf * 8);
            a1.u[0] = *(const uint4*)(ar1 + khalf * 8);
            a1.u[1] = *(const uint4*)(ar1 + 16 + khalf * 8);
            const __bf16* br0 = Bb + (wn * 32 + l15)      * LDS_K + kc * 32 + khalf * 8;
            const __bf16* br1 = Bb + (wn * 32 + 16 + l15) * LDS_K + kc * 32 + khalf * 8;
            b0.u[0] = *(const uint4*)(br0);
            b0.u[1] = *(const uint4*)(br0 + 16);
            b1.u[0] = *(const uint4*)(br1);
            b1.u[1] = *(const uint4*)(br1 + 16);
            acc[0][0] = WMMA_BF16(a0.v, b0.v, acc[0][0]);
            acc[0][1] = WMMA_BF16(a0.v, b1.v, acc[0][1]);
            acc[1][0] = WMMA_BF16(a1.v, b0.v, acc[1][0]);
            acc[1][1] = WMMA_BF16(a1.v, b1.v, acc[1][1]);
        }
        wait_async0();       // own prefetch done before anyone flips buffers
        __syncthreads();     // everyone's prefetch done + all reads of buf p done
    }
}

// Kernel 1: QKV projection. grid = (16, 64, 3).
// Q -> [B,H,T,DH] bf16, PRE-SCALED by 1/sqrt(DH)=0.125 (exact in bf16).
// K -> [B,H,T,DH] bf16.  V -> [B,H,DH,T] bf16 (pre-transposed for attention).
__global__ __launch_bounds__(256) void gemm_qkv(const __bf16* __restrict__ xb,
                                                const __bf16* __restrict__ wqT,
                                                const __bf16* __restrict__ wkT,
                                                const __bf16* __restrict__ wvT,
                                                __bf16* __restrict__ qo,
                                                __bf16* __restrict__ ko,
                                                __bf16* __restrict__ vo) {
    __shared__ __bf16 As[2 * AS_BUF];
    __shared__ __bf16 Bs[2 * BS_BUF];

    const int which = blockIdx.z;
    const __bf16* Wt = (which == 0) ? wqT : (which == 1) ? wkT : wvT;
    __bf16* O        = (which == 0) ? qo  : (which == 1) ? ko  : vo;
    const float oscale = (which == 0) ? 0.125f : 1.0f;

    const int m0 = blockIdx.y * 128, n0 = blockIdx.x * 64;
    v8f acc[2][2] = {};
    gemm_core(xb, Wt, As, Bs, m0, n0, acc);

    const int lane = threadIdx.x & 31, wid = threadIdx.x >> 5;
    const int wm = wid >> 1, wn = wid & 1;
    const int l15 = lane & 15, mhalf = lane >> 4;
    #pragma unroll
    for (int mt = 0; mt < 2; ++mt) {
        #pragma unroll
        for (int nt = 0; nt < 2; ++nt) {
            #pragma unroll
            for (int r = 0; r < 8; ++r) {
                const int gm = m0 + wm * 32 + mt * 16 + r + 8 * mhalf;
                const int b  = gm >> 11, t = gm & (TT - 1);
                const int c  = n0 + wn * 32 + nt * 16 + l15;
                const int h  = c >> 6, dh = c & 63;
                const __bf16 val = (__bf16)(acc[mt][nt][r] * oscale);
                if (which == 2) {
                    O[(((size_t)(b * HH + h)) * DH + dh) * TT + t] = val;   // [B,H,DH,T]
                } else {
                    O[(((size_t)(b * HH + h)) * TT + t) * DH + dh] = val;   // [B,H,T,DH]
                }
            }
        }
    }
}

// Kernel 3: output projection + bias -> fp32. grid = (16, 64).
__global__ __launch_bounds__(256) void gemm_out(const __bf16* __restrict__ ctx,
                                                const __bf16* __restrict__ woT,
                                                const float* __restrict__ bo,
                                                float* __restrict__ out) {
    __shared__ __bf16 As[2 * AS_BUF];
    __shared__ __bf16 Bs[2 * BS_BUF];

    const int m0 = blockIdx.y * 128, n0 = blockIdx.x * 64;
    v8f acc[2][2] = {};
    gemm_core(ctx, woT, As, Bs, m0, n0, acc);

    const int lane = threadIdx.x & 31, wid = threadIdx.x >> 5;
    const int wm = wid >> 1, wn = wid & 1;
    const int l15 = lane & 15, mhalf = lane >> 4;
    #pragma unroll
    for (int mt = 0; mt < 2; ++mt) {
        #pragma unroll
        for (int nt = 0; nt < 2; ++nt) {
            #pragma unroll
            for (int r = 0; r < 8; ++r) {
                const int gm = m0 + wm * 32 + mt * 16 + r + 8 * mhalf;
                const int c  = n0 + wn * 32 + nt * 16 + l15;
                out[(size_t)gm * DD + c] = acc[mt][nt][r] + bo[c];
            }
        }
    }
}

// ---------------------------------------------------------------------------
// Kernel 2: causal flash attention. grid = (T/64, B*H), 128 threads (4 waves).
// Q pre-scaled by 0.125. K [B,H,T,DH], V [B,H,DH,T]. K/V tiles staged by the
// TDM, double-buffered (TDM in-order => wait_tensorcnt(2) = current tile done).
// Writes ctx in [B,T,D] bf16 layout.
// ---------------------------------------------------------------------------
__global__ __launch_bounds__(128) void attn_flash(const __bf16* __restrict__ Qb,
                                                  const __bf16* __restrict__ Kb,
                                                  const __bf16* __restrict__ Vb,
                                                  __bf16* __restrict__ Ctx) {
    __shared__ __bf16 Ks[2][64 * LDS_K];     // [key][dh]
    __shared__ __bf16 Vt[2][64 * LDS_K];     // [dh][key]
    __shared__ __bf16 Ps[4 * 16 * LDS_K];    // per-wave P tile [16][64]

    const int q0   = blockIdx.x * 64;
    const int bh   = blockIdx.y;
    const int tid  = threadIdx.x;
    const int lane = tid & 31, wid = tid >> 5;
    const int khalf = lane >> 4, l15 = lane & 15;

    const __bf16* Qg = Qb + (size_t)bh * TT * DH;
    const __bf16* Kg = Kb + (size_t)bh * TT * DH;
    const __bf16* Vg = Vb + (size_t)bh * DH * TT;

    // Preload Q fragments for this wave's 16 rows (K-dim 64 -> 2 chunks)
    FragBF qa0, qa1;
    {
        const __bf16* qp = Qg + (size_t)(q0 + wid * 16 + l15) * DH;
        qa0.u[0] = *(const uint4*)(qp + khalf * 8);
        qa0.u[1] = *(const uint4*)(qp + 16 + khalf * 8);
        qa1.u[0] = *(const uint4*)(qp + 32 + khalf * 8);
        qa1.u[1] = *(const uint4*)(qp + 48 + khalf * 8);
    }

    v8f c0 = {}, c1 = {}, c2 = {}, c3 = {};
    float mrow[8], lrow[8];
    #pragma unroll
    for (int r = 0; r < 8; ++r) { mrow[r] = -__builtin_inff(); lrow[r] = 0.f; }

    // Prologue: TDM-stage tile 0 into buffer 0
    if (wid == 0) {
        tdm_load_2d_64x64((unsigned)(size_t)(void*)Ks[0],
                          (unsigned long long)(size_t)Kg, DH, TT, DH);
        tdm_load_2d_64x64((unsigned)(size_t)(void*)Vt[0],
                          (unsigned long long)(size_t)Vg, TT, DH, TT);
    }

    for (int kt0 = 0; kt0 <= q0 + 63; kt0 += 64) {
        const int p = (kt0 >> 6) & 1;
        const bool more = (kt0 + 64) <= (q0 + 63);
        if (wid == 0) {
            if (more) {      // prefetch next tile into other buffer, then wait
                const int kt1 = kt0 + 64;
                tdm_load_2d_64x64((unsigned)(size_t)(void*)Ks[p ^ 1],
                                  (unsigned long long)(size_t)(Kg + (size_t)kt1 * DH),
                                  DH, TT, DH);
                tdm_load_2d_64x64((unsigned)(size_t)(void*)Vt[p ^ 1],
                                  (unsigned long long)(size_t)(Vg + kt1),
                                  TT, DH, TT);
                __builtin_amdgcn_s_wait_tensorcnt(2);   // current tile landed
            } else {
                __builtin_amdgcn_s_wait_tensorcnt(0);
            }
        }
        __syncthreads();

        // S = Q * K^T  (16 queries x 64 keys) -- Q carries the 1/8 scale
        v8f s0 = {}, s1 = {}, s2 = {}, s3 = {};
        #pragma unroll
        for (int kc = 0; kc < 2; ++kc) {
            const v16bf a = kc ? qa1.v : qa0.v;
            #pragma unroll
            for (int nt = 0; nt < 4; ++nt) {
                FragBF b;
                const __bf16* kr = Ks[p] + (nt * 16 + l15) * LDS_K + kc * 32 + khalf * 8;
                b.u[0] = *(const uint4*)(kr);
                b.u[1] = *(const uint4*)(kr + 16);
                v8f* sp = (nt == 0) ? &s0 : (nt == 1) ? &s1 : (nt == 2) ? &s2 : &s3;
                *sp = WMMA_BF16(a, b.v, *sp);
            }
        }

        // causal mask, only on the diagonal band (wave-uniform test)
        if (kt0 + 63 > q0 + wid * 16) {
            #pragma unroll
            for (int r = 0; r < 8; ++r) {
                const int q = q0 + wid * 16 + r + 8 * khalf;
                #pragma unroll
                for (int nt = 0; nt < 4; ++nt) {
                    v8f* sp = (nt == 0) ? &s0 : (nt == 1) ? &s1 : (nt == 2) ? &s2 : &s3;
                    const int key = kt0 + nt * 16 + l15;
                    if (key > q) (*sp)[r] = -__builtin_inff();
                }
            }
        }

        // online softmax update; P -> LDS in A-fragment layout
        #pragma unroll
        for (int r = 0; r < 8; ++r) {
            float mx = fmaxf(fmaxf(s0[r], s1[r]), fmaxf(s2[r], s3[r]));
            mx = rowmax16(mx);
            const float mnew = fmaxf(mrow[r], mx);
            const float corr = __expf(mrow[r] - mnew);
            const float p0 = __expf(s0[r] - mnew);
            const float p1 = __expf(s1[r] - mnew);
            const float p2 = __expf(s2[r] - mnew);
            const float p3 = __expf(s3[r] - mnew);
            const float ps = rowsum16(p0 + p1 + p2 + p3);
            lrow[r] = lrow[r] * corr + ps;
            mrow[r] = mnew;
            c0[r] *= corr; c1[r] *= corr; c2[r] *= corr; c3[r] *= corr;

            __bf16* prow = Ps + (wid * 16 + r + 8 * khalf) * LDS_K;
            prow[ 0 + l15] = (__bf16)p0;
            prow[16 + l15] = (__bf16)p1;
            prow[32 + l15] = (__bf16)p2;
            prow[48 + l15] = (__bf16)p3;
        }

        // ctx += P * V
        #pragma unroll
        for (int kc = 0; kc < 2; ++kc) {
            FragBF pa;
            const __bf16* pr = Ps + (wid * 16 + l15) * LDS_K + kc * 32;
            pa.u[0] = *(const uint4*)(pr + khalf * 8);
            pa.u[1] = *(const uint4*)(pr + 16 + khalf * 8);
            #pragma unroll
            for (int nt = 0; nt < 4; ++nt) {
                FragBF b;
                const __bf16* vr = Vt[p] + (nt * 16 + l15) * LDS_K + kc * 32 + khalf * 8;
                b.u[0] = *(const uint4*)(vr);
                b.u[1] = *(const uint4*)(vr + 16);
                v8f* cp = (nt == 0) ? &c0 : (nt == 1) ? &c1 : (nt == 2) ? &c2 : &c3;
                *cp = WMMA_BF16(pa.v, b.v, *cp);
            }
        }
        __syncthreads();    // all reads of buffer p done before it is re-staged
    }

    // normalize and store ctx in [B,T,D] bf16
    const int b = bh >> 4, h = bh & 15;
    #pragma unroll
    for (int r = 0; r < 8; ++r) {
        const float inv = 1.0f / lrow[r];
        const int tq = q0 + wid * 16 + r + 8 * khalf;
        __bf16* orow = Ctx + ((size_t)(b * TT + tq)) * DD + h * DH;
        orow[ 0 + l15] = (__bf16)(c0[r] * inv);
        orow[16 + l15] = (__bf16)(c1[r] * inv);
        orow[32 + l15] = (__bf16)(c2[r] * inv);
        orow[48 + l15] = (__bf16)(c3[r] * inv);
    }
}

// ---------------------------------------------------------------------------
// Launch
// ---------------------------------------------------------------------------
extern "C" void kernel_launch(void* const* d_in, const int* in_sizes, int n_in,
                              void* d_out, int out_size, void* d_ws, size_t ws_size,
                              hipStream_t stream) {
    const float* x  = (const float*)d_in[0];
    const float* Wq = (const float*)d_in[1];
    const float* Wk = (const float*)d_in[2];
    const float* Wv = (const float*)d_in[3];
    const float* Wo = (const float*)d_in[4];
    const float* bo = (const float*)d_in[5];
    float* out = (float*)d_out;

    char* ws = (char*)d_ws;
    size_t off = 0;
    __bf16* xb  = (__bf16*)(ws + off); off += (size_t)MM * DD * 2;
    __bf16* wqT = (__bf16*)(ws + off); off += (size_t)DD * DD * 2;
    __bf16* wkT = (__bf16*)(ws + off); off += (size_t)DD * DD * 2;
    __bf16* wvT = (__bf16*)(ws + off); off += (size_t)DD * DD * 2;
    __bf16* woT = (__bf16*)(ws + off); off += (size_t)DD * DD * 2;
    __bf16* qb  = (__bf16*)(ws + off); off += (size_t)MM * DD * 2;
    __bf16* kb  = (__bf16*)(ws + off); off += (size_t)MM * DD * 2;
    __bf16* vb  = (__bf16*)(ws + off); off += (size_t)MM * DD * 2;
    __bf16* ctx = (__bf16*)(ws + off); off += (size_t)MM * DD * 2;

    const int nx = MM * DD;
    cvt_f32_bf16<<<(nx / 4 + 255) / 256, 256, 0, stream>>>(x, xb, nx);
    dim3 tg(DD / 16, DD / 16);
    cvt_transpose<<<tg, 256, 0, stream>>>(Wq, wqT);
    cvt_transpose<<<tg, 256, 0, stream>>>(Wk, wkT);
    cvt_transpose<<<tg, 256, 0, stream>>>(Wv, wvT);
    cvt_transpose<<<tg, 256, 0, stream>>>(Wo, woT);

    gemm_qkv<<<dim3(DD / 64, MM / 128, 3), 256, 0, stream>>>(xb, wqT, wkT, wvT, qb, kb, vb);
    attn_flash<<<dim3(TT / 64, BB * HH), 128, 0, stream>>>(qb, kb, vb, ctx);
    gemm_out<<<dim3(DD / 64, MM / 128), 256, 0, stream>>>(ctx, woT, bo, out);
}
